// NMPEUModel_44590350467106
// MI455X (gfx1250) — matrix operand order repaired and checked by reading, not compiled, not run on hardware
//
#include <hip/hip_runtime.h>
#include <hip/hip_bf16.h>

// ---------------------------------------------------------------------------
// NMP-EU (SchNet-like) message passing net for MI455X (gfx1250).
// All dense per-edge / per-node MLPs run on WMMA f16 (f32 accumulate).
// Weights are pre-swizzled once into the WMMA B-fragment lane layout.
// ---------------------------------------------------------------------------

#define DIMV   64
#define RBFV   64
#define LN2F   0.6931471805599453f
#define CUTOFF 5.0f

typedef _Float16 half16 __attribute__((ext_vector_type(16)));
typedef float    float8 __attribute__((ext_vector_type(8)));

union AFrag { half16 v; unsigned int u[8]; };

__device__ __forceinline__ float sspf(float x) {
    // shifted softplus: softplus(x) - ln2, numerically stable
    float ax = fabsf(x);
    return fmaxf(x, 0.0f) + __logf(1.0f + __expf(-ax)) - LN2F;
}

__device__ __forceinline__ unsigned int pack2h(float a, float b) {
    union { _Float16 h[2]; unsigned int u; } p;
    p.h[0] = (_Float16)a; p.h[1] = (_Float16)b;
    return p.u;
}

__device__ __forceinline__ void atomicAddF(float* p, float v) {
    (void)__hip_atomic_fetch_add(p, v, __ATOMIC_RELAXED, __HIP_MEMORY_SCOPE_AGENT);
}

// A fragment (16x32 f16) from an LDS tile stored row-major with strideH halfs.
// ISA layout (CDNA5 ISA): lanes 0-15: VGPR0..3 K=0..7, VGPR4..7 K=16..23;
// lanes 16-31: K=8..15 / K=24..31; row M = lane & 15.
__device__ __forceinline__ half16 load_a_frag(const _Float16* tile, int strideH, int kb) {
    const int l  = threadIdx.x & 31;
    const int m  = l & 15;
    const int hf = l >> 4;
    AFrag f;
#pragma unroll
    for (int i = 0; i < 8; ++i) {
        int k0 = kb * 32 + (i & 3) * 2 + hf * 8 + (i >> 2) * 16;
        f.u[i] = *(const unsigned int*)(tile + m * strideH + k0);
    }
    return f.v;
}

// B fragment (32x16 f16) from pre-swizzled weights: each lane's 16 halfs contiguous.
__device__ __forceinline__ half16 load_b_frag(const _Float16* swz, int fragIdx) {
    const int l = threadIdx.x & 31;
    return *(const half16*)(swz + ((size_t)fragIdx * 32 + l) * 16);
}

__device__ __forceinline__ float8 wmma16(half16 a, half16 b, float8 c) {
    return __builtin_amdgcn_wmma_f32_16x16x32_f16(false, a, false, b, (short)0, c, false, false);
}

// Store f32 C fragment (16x16) into an LDS f16 tile (row-major, strideH halfs).
__device__ __forceinline__ void store_c_frag_f16(_Float16* tile, int strideH, int nb, float8 c) {
    const int l  = threadIdx.x & 31;
    const int n  = nb * 16 + (l & 15);
    const int hf = l >> 4;
#pragma unroll
    for (int r = 0; r < 8; ++r)
        tile[(r + hf * 8) * strideH + n] = (_Float16)c[r];
}

// ---------------------------------------------------------------------------
// Init kernels
// ---------------------------------------------------------------------------
__global__ void embed_kernel(const int* __restrict__ z, const float* __restrict__ emb,
                             float* __restrict__ h, int N) {
    int i = blockIdx.x * blockDim.x + threadIdx.x;
    if (i >= N * DIMV) return;
    int node = i >> 6, d = i & 63;
    h[i] = emb[z[node] * DIMV + d];
}

__global__ void rbf_kernel(const float* __restrict__ dist, float* __restrict__ e, int E) {
    int i = blockIdx.x * blockDim.x + threadIdx.x;
    if (i >= E * RBFV) return;
    int ei = i >> 6, j = i & 63;
    const float gap = CUTOFF / (float)(RBFV - 1);
    float diff = dist[ei] - (float)j * gap;
    e[i] = __expf(-(diff * diff) / gap);
}

__global__ void zero_kernel(float* __restrict__ p, int n) {
    int i = blockIdx.x * blockDim.x + threadIdx.x;
    if (i < n) p[i] = 0.0f;
}

// Swizzle a row-major f32 weight W[K x N] into WMMA B-fragment order, f16.
// dst element t: frag = t/512 (= nb*(K/32)+kb), lane = (t%512)/16, j = t%16,
// source k = kb*32 + (j>>3)*16 + (lane>>4)*8 + (j&7), n = nb*16 + (lane&15).
__global__ void swizzle_kernel(const float* __restrict__ W, _Float16* __restrict__ out,
                               int K, int N) {
    int t = blockIdx.x * blockDim.x + threadIdx.x;
    int total = K * N;
    if (t >= total) return;
    int KB   = K >> 5;
    int frag = t >> 9;
    int rem  = t & 511;
    int lane = rem >> 4;
    int j    = rem & 15;
    int nb   = frag / KB;
    int kb   = frag - nb * KB;
    int n    = nb * 16 + (lane & 15);
    int k    = kb * 32 + (j >> 3) * 16 + (lane >> 4) * 8 + (j & 7);
    out[t] = (_Float16)W[k * N + n];
}

// ---------------------------------------------------------------------------
// Edge kernel: per 16-edge tile (one wave each), fused:
//   cat = [h[src], h[dst], e]  (16 x 192)
//   e'  = ssp(cat @ W1 + b1) @ W2 + b2                (written back to e)
//   he  = ssp(e' @ PE1 + pb1) @ PE2 + pb2             (scatter-add into agg[dst])
// ---------------------------------------------------------------------------
__global__ __launch_bounds__(128) void edge_layer_kernel(
    const float* __restrict__ h, float* __restrict__ e,
    const int* __restrict__ src, const int* __restrict__ dst,
    float* __restrict__ agg,
    const _Float16* __restrict__ w1s, const float* __restrict__ b1,
    const _Float16* __restrict__ w2s, const float* __restrict__ b2,
    const _Float16* __restrict__ pe1s, const float* __restrict__ pb1,
    const _Float16* __restrict__ pe2s, const float* __restrict__ pb2,
    int nTiles, int E)
{
    __shared__ _Float16 smem[4 * 6144];
    const int wave = threadIdx.x >> 5;
    const int lane = threadIdx.x & 31;
    const int tile = blockIdx.x * 4 + wave;
    if (tile >= nTiles) return;

    _Float16* catA = smem + wave * 6144;   // 16 x 192
    _Float16* act1 = catA + 3072;          // 16 x 128
    _Float16* act2 = act1 + 2048;          // 16 x 64
    _Float16* act3 = catA;                 // reuse (16 x 64), catA dead after GEMM1

    const int e0 = tile * 16;

    __builtin_prefetch(w1s, 0, 0);
    __builtin_prefetch(w2s, 0, 0);

    // ---- gather concat tile into LDS (f32 -> f16) -------------------------
#pragma unroll 4
    for (int r = 0; r < 16; ++r) {
        int ei = e0 + r; if (ei > E - 1) ei = E - 1;
        int s = src[ei], d = dst[ei];
        float2 hs = *(const float2*)(h + (size_t)s * DIMV + 2 * lane);
        float2 hd = *(const float2*)(h + (size_t)d * DIMV + 2 * lane);
        float2 ee = *(const float2*)(e + (size_t)ei * RBFV + 2 * lane);
        *(unsigned int*)(catA + r * 192 +       2 * lane) = pack2h(hs.x, hs.y);
        *(unsigned int*)(catA + r * 192 +  64 + 2 * lane) = pack2h(hd.x, hd.y);
        *(unsigned int*)(catA + r * 192 + 128 + 2 * lane) = pack2h(ee.x, ee.y);
    }

    // ---- GEMM1: (16x192) @ (192x128), ssp ---------------------------------
#pragma unroll
    for (int nb = 0; nb < 8; ++nb) {
        float8 acc = {};
#pragma unroll
        for (int kb = 0; kb < 6; ++kb)
            acc = wmma16(load_a_frag(catA, 192, kb), load_b_frag(w1s, nb * 6 + kb), acc);
        float bv = b1[nb * 16 + (lane & 15)];
#pragma unroll
        for (int r = 0; r < 8; ++r) acc[r] = sspf(acc[r] + bv);
        store_c_frag_f16(act1, 128, nb, acc);
    }

    // ---- GEMM2: (16x128) @ (128x64) -> new edge features ------------------
    {
        const int n  = lane & 15;
        const int hf = lane >> 4;
#pragma unroll
        for (int nb = 0; nb < 4; ++nb) {
            float8 acc = {};
#pragma unroll
            for (int kb = 0; kb < 4; ++kb)
                acc = wmma16(load_a_frag(act1, 128, kb), load_b_frag(w2s, nb * 4 + kb), acc);
            float bv = b2[nb * 16 + n];
#pragma unroll
            for (int r = 0; r < 8; ++r) {
                int row = r + hf * 8;
                float v = acc[r] + bv;
                if (e0 + row < E)
                    e[(size_t)(e0 + row) * RBFV + nb * 16 + n] = v;  // in-place update
                act2[row * 64 + nb * 16 + n] = (_Float16)v;
            }
        }
    }

    // ---- GEMM3: project_edge stage 1 (64x64), ssp -------------------------
#pragma unroll
    for (int nb = 0; nb < 4; ++nb) {
        float8 acc = {};
#pragma unroll
        for (int kb = 0; kb < 2; ++kb)
            acc = wmma16(load_a_frag(act2, 64, kb), load_b_frag(pe1s, nb * 2 + kb), acc);
        float bv = pb1[nb * 16 + (lane & 15)];
#pragma unroll
        for (int r = 0; r < 8; ++r) acc[r] = sspf(acc[r] + bv);
        store_c_frag_f16(act3, 64, nb, acc);
    }

    // ---- GEMM4: project_edge stage 2 (64x64), scatter-add to agg[dst] -----
    {
        const int n  = lane & 15;
        const int hf = lane >> 4;
#pragma unroll
        for (int nb = 0; nb < 4; ++nb) {
            float8 acc = {};
#pragma unroll
            for (int kb = 0; kb < 2; ++kb)
                acc = wmma16(load_a_frag(act3, 64, kb), load_b_frag(pe2s, nb * 2 + kb), acc);
            float bv = pb2[nb * 16 + n];
#pragma unroll
            for (int r = 0; r < 8; ++r) {
                int row = r + hf * 8;
                int ei  = e0 + row;
                if (ei < E) {
                    int node = dst[ei];
                    atomicAddF(agg + (size_t)node * DIMV + nb * 16 + n, acc[r] + bv);
                }
            }
        }
    }
}

// ---------------------------------------------------------------------------
// Node kernel: new = ssp(agg @ A + ab) @ B + bb ; h += new  (16-node tiles)
// ---------------------------------------------------------------------------
__global__ __launch_bounds__(128) void node_layer_kernel(
    float* __restrict__ h, const float* __restrict__ agg,
    const _Float16* __restrict__ as, const float* __restrict__ ab,
    const _Float16* __restrict__ bs, const float* __restrict__ bb,
    int nTiles, int N)
{
    __shared__ _Float16 smem[4 * 2048];
    const int wave = threadIdx.x >> 5;
    const int lane = threadIdx.x & 31;
    const int tile = blockIdx.x * 4 + wave;
    if (tile >= nTiles) return;

    _Float16* A   = smem + wave * 2048;   // 16 x 64
    _Float16* act = A + 1024;             // 16 x 64
    const int n0 = tile * 16;

#pragma unroll 4
    for (int r = 0; r < 16; ++r) {
        int ni = n0 + r; if (ni > N - 1) ni = N - 1;
        float2 v = *(const float2*)(agg + (size_t)ni * DIMV + 2 * lane);
        *(unsigned int*)(A + r * 64 + 2 * lane) = pack2h(v.x, v.y);
    }

#pragma unroll
    for (int nb = 0; nb < 4; ++nb) {
        float8 acc = {};
#pragma unroll
        for (int kb = 0; kb < 2; ++kb)
            acc = wmma16(load_a_frag(A, 64, kb), load_b_frag(as, nb * 2 + kb), acc);
        float bv = ab[nb * 16 + (lane & 15)];
#pragma unroll
        for (int r = 0; r < 8; ++r) acc[r] = sspf(acc[r] + bv);
        store_c_frag_f16(act, 64, nb, acc);
    }

    {
        const int n  = lane & 15;
        const int hf = lane >> 4;
#pragma unroll
        for (int nb = 0; nb < 4; ++nb) {
            float8 acc = {};
#pragma unroll
            for (int kb = 0; kb < 2; ++kb)
                acc = wmma16(load_a_frag(act, 64, kb), load_b_frag(bs, nb * 2 + kb), acc);
            float bv = bb[nb * 16 + n];
#pragma unroll
            for (int r = 0; r < 8; ++r) {
                int row = n0 + r + hf * 8;
                if (row < N) {
                    size_t idx = (size_t)row * DIMV + nb * 16 + n;
                    h[idx] = h[idx] + acc[r] + bv;    // residual
                }
            }
        }
    }
}

// ---------------------------------------------------------------------------
// Readout: atom = ssp(h @ d1 + b1); res = atom @ d2 + b2; per-graph mean accum
// ---------------------------------------------------------------------------
__global__ __launch_bounds__(128) void readout_kernel(
    const float* __restrict__ h, const int* __restrict__ gids,
    const _Float16* __restrict__ d1s, const float* __restrict__ d1b,
    const float* __restrict__ d2w, const float* __restrict__ d2b,
    float* __restrict__ sums, float* __restrict__ cnt,
    int nTiles, int N)
{
    __shared__ _Float16 aT[4][1024];      // 16 x 64 f16 per wave
    __shared__ float    atomT[4][512];    // 16 x 32 f32 per wave
    const int wave = threadIdx.x >> 5;
    const int lane = threadIdx.x & 31;
    const int tile = blockIdx.x * 4 + wave;
    if (tile >= nTiles) return;
    const int n0 = tile * 16;

    _Float16* A = aT[wave];
#pragma unroll 4
    for (int r = 0; r < 16; ++r) {
        int ni = n0 + r; if (ni > N - 1) ni = N - 1;
        float2 v = *(const float2*)(h + (size_t)ni * DIMV + 2 * lane);
        *(unsigned int*)(A + r * 64 + 2 * lane) = pack2h(v.x, v.y);
    }

#pragma unroll
    for (int nb = 0; nb < 2; ++nb) {
        float8 acc = {};
#pragma unroll
        for (int kb = 0; kb < 2; ++kb)
            acc = wmma16(load_a_frag(A, 64, kb), load_b_frag(d1s, nb * 2 + kb), acc);
        float bv = d1b[nb * 16 + (lane & 15)];
        const int n  = nb * 16 + (lane & 15);
        const int hf = lane >> 4;
#pragma unroll
        for (int r = 0; r < 8; ++r)
            atomT[wave][(r + hf * 8) * 32 + n] = sspf(acc[r] + bv);
    }

    if (lane < 16) {
        int ni = n0 + lane;
        if (ni < N) {
            float dot = d2b[0];
#pragma unroll
            for (int j = 0; j < 32; ++j) dot += atomT[wave][lane * 32 + j] * d2w[j];
            int g = gids[ni];
            atomicAddF(sums + g, dot);
            atomicAddF(cnt + g, 1.0f);
        }
    }
}

__global__ void finalize_kernel(const float* __restrict__ sums, const float* __restrict__ cnt,
                                float* __restrict__ out, int G) {
    int g = blockIdx.x * blockDim.x + threadIdx.x;
    if (g < G) out[g] = sums[g] / fmaxf(cnt[g], 1.0f);
}

// ---------------------------------------------------------------------------
// Host launcher
// ---------------------------------------------------------------------------
extern "C" void kernel_launch(void* const* d_in, const int* in_sizes, int n_in,
                              void* d_out, int out_size, void* d_ws, size_t ws_size,
                              hipStream_t stream) {
    (void)n_in; (void)ws_size;
    const int N = in_sizes[0];          // nodes
    const int E = in_sizes[1];          // edges
    const int L = 3;                    // conv layers
    const int G = out_size;             // graphs

    const int*   atom_z = (const int*)d_in[0];
    const float* dist   = (const float*)d_in[1];
    const int*   src    = (const int*)d_in[2];
    const int*   dst    = (const int*)d_in[3];
    const int*   gids   = (const int*)d_in[4];
    const float* emb    = (const float*)d_in[5];
    const float* eu_w1  = (const float*)d_in[6];
    const float* eu_b1  = (const float*)d_in[7];
    const float* eu_w2  = (const float*)d_in[8];
    const float* eu_b2  = (const float*)d_in[9];
    const float* pe1_w  = (const float*)d_in[10];
    const float* pe1_b  = (const float*)d_in[11];
    const float* pe2_w  = (const float*)d_in[12];
    const float* pe2_b  = (const float*)d_in[13];
    const float* pn2a_w = (const float*)d_in[14];
    const float* pn2a_b = (const float*)d_in[15];
    const float* pn2b_w = (const float*)d_in[16];
    const float* pn2b_b = (const float*)d_in[17];
    const float* d1_w   = (const float*)d_in[18];
    const float* d1_b   = (const float*)d_in[19];
    const float* d2_w   = (const float*)d_in[20];
    const float* d2_b   = (const float*)d_in[21];

    // workspace carve-out (256B aligned chunks)
    size_t off = 0;
    auto take = [&](size_t bytes) -> char* {
        char* p = (char*)d_ws + off;
        off += (bytes + 255) & ~(size_t)255;
        return p;
    };
    float*     h    = (float*)take((size_t)N * DIMV * sizeof(float));
    float*     e    = (float*)take((size_t)E * RBFV * sizeof(float));
    float*     agg  = (float*)take((size_t)N * DIMV * sizeof(float));
    float*     sums = (float*)take((size_t)G * sizeof(float));
    float*     cnt  = (float*)take((size_t)G * sizeof(float));
    _Float16*  wswz = (_Float16*)take((size_t)(L * (192*128 + 128*64 + 4*64*64) + 64*32)
                                      * sizeof(_Float16));

    // per-layer swizzled weight pointers
    _Float16 *w1s[3], *w2s[3], *pe1s[3], *pe2s[3], *pn2as[3], *pn2bs[3];
    _Float16* p = wswz;
    for (int l = 0; l < L; ++l) {
        w1s[l]   = p; p += 192 * 128;
        w2s[l]   = p; p += 128 * 64;
        pe1s[l]  = p; p += 64 * 64;
        pe2s[l]  = p; p += 64 * 64;
        pn2as[l] = p; p += 64 * 64;
        pn2bs[l] = p; p += 64 * 64;
    }
    _Float16* d1s = p;

    const int T = 256;
    auto blocks = [](long n, int t) { return (int)((n + t - 1) / t); };

    // weight swizzle (f32 row-major -> f16 WMMA B fragments)
    for (int l = 0; l < L; ++l) {
        swizzle_kernel<<<blocks(192*128, T), T, 0, stream>>>(eu_w1  + (size_t)l*192*128, w1s[l],   192, 128);
        swizzle_kernel<<<blocks(128*64,  T), T, 0, stream>>>(eu_w2  + (size_t)l*128*64,  w2s[l],   128, 64);
        swizzle_kernel<<<blocks(64*64,   T), T, 0, stream>>>(pe1_w  + (size_t)l*64*64,   pe1s[l],  64, 64);
        swizzle_kernel<<<blocks(64*64,   T), T, 0, stream>>>(pe2_w  + (size_t)l*64*64,   pe2s[l],  64, 64);
        swizzle_kernel<<<blocks(64*64,   T), T, 0, stream>>>(pn2a_w + (size_t)l*64*64,   pn2as[l], 64, 64);
        swizzle_kernel<<<blocks(64*64,   T), T, 0, stream>>>(pn2b_w + (size_t)l*64*64,   pn2bs[l], 64, 64);
    }
    swizzle_kernel<<<blocks(64*32, T), T, 0, stream>>>(d1_w, d1s, 64, 32);

    // init node/edge features
    embed_kernel<<<blocks((long)N * DIMV, T), T, 0, stream>>>(atom_z, emb, h, N);
    rbf_kernel<<<blocks((long)E * RBFV, T), T, 0, stream>>>(dist, e, E);

    const int ET = (E + 15) / 16;
    const int NT = (N + 15) / 16;

    for (int l = 0; l < L; ++l) {
        zero_kernel<<<blocks((long)N * DIMV, T), T, 0, stream>>>(agg, N * DIMV);
        edge_layer_kernel<<<(ET + 3) / 4, 128, 0, stream>>>(
            h, e, src, dst, agg,
            w1s[l],  eu_b1  + (size_t)l * 128,
            w2s[l],  eu_b2  + (size_t)l * 64,
            pe1s[l], pe1_b  + (size_t)l * 64,
            pe2s[l], pe2_b  + (size_t)l * 64,
            ET, E);
        node_layer_kernel<<<(NT + 3) / 4, 128, 0, stream>>>(
            h, agg,
            pn2as[l], pn2a_b + (size_t)l * 64,
            pn2bs[l], pn2b_b + (size_t)l * 64,
            NT, N);
    }

    zero_kernel<<<blocks(2L * G, T), T, 0, stream>>>(sums, 2 * G);  // cnt starts after sums' 256B padding; fully zeroed only if G % 64 == 0
    readout_kernel<<<(NT + 3) / 4, 128, 0, stream>>>(
        h, gids, d1s, d1_b, d2_w, d2_b, sums, cnt, NT, N);
    finalize_kernel<<<blocks(G, T), T, 0, stream>>>(sums, cnt, (float*)d_out, G);
}
